// E3nnModel_12515534700917
// MI455X (gfx1250) — matrix operand order repaired and checked by reading, not compiled, not run on hardware
//
#include <hip/hip_runtime.h>
#include <cmath>

// ---------------------------------------------------------------------------
// MI455X (gfx1250, wave32) implementation of the e3nn-style GNN reference.
//
//  * All dense math (embeddings, radial MLP, every fctp tensor product) runs
//    on v_wmma_f32_16x16x32_f16, f32 accumulation.
//  * GEMM operands are pre-cast to zero-padded f16 so the inner loop is pure
//    global_load_b128 + v_wmma (no guarded scalar loads).
//  * fctp(out[n,w,i] = sum_{u,a} x[n,u,i] attr[n,a] W[u,a,w]) is a GEMM with
//    K = mu*64 whose A fragment is the outer product x_i (x) attr, built per
//    16-node tile from LDS-staged x/attr tiles.
//  * Edge message passing (gather + CG contraction + radial weight + scatter)
//    is one fused kernel per conv layer, scatter via global f32 atomics.
// ---------------------------------------------------------------------------

typedef __attribute__((ext_vector_type(16))) _Float16 v16h;
typedef __attribute__((ext_vector_type(8)))  _Float16 v8h;
typedef __attribute__((ext_vector_type(8)))  float    v8f;

__device__ __forceinline__ float sigm(float x) { return 1.f / (1.f + __expf(-x)); }

// ---- weight cast: W f32 (K,M) row-major -> Wt f16 (Mp,Kp), zero padded -----
__global__ void k_cast_wt(const float* __restrict__ W, _Float16* __restrict__ Wt,
                          int K, int M, int Kp, int Mp) {
  int idx = blockIdx.x * 256 + threadIdx.x;
  if (idx >= Mp * Kp) return;
  int m = idx / Kp, k = idx - m * Kp;
  float v = (m < M && k < K) ? W[(size_t)k * M + m] : 0.f;
  Wt[idx] = (_Float16)v;
}

// ---- activation cast: X f32 (n,Kx) ld=ldx -> Xh f16 (n,Kp), zero padded ----
__global__ void k_cast_xh(const float* __restrict__ X, int ldx, int Kx,
                          _Float16* __restrict__ Xh, int Kp, int n) {
  int idx = blockIdx.x * 256 + threadIdx.x;
  if (idx >= n * Kp) return;
  int row = idx / Kp, k = idx - row * Kp;
  Xh[idx] = (_Float16)((k < Kx) ? X[(size_t)row * ldx + k] : 0.f);
}

// ---- WMMA GEMM: act(Xh[n,Kp] * Wt^T + b) -> Y (f32) or Yh (f16) ------------
// One wave computes one 16x16 tile; inner loop is 4x b128 loads + 1 wmma.
__global__ void k_gemm(const _Float16* __restrict__ Xh, int ldx,
                       const _Float16* __restrict__ Wt, int Kp,
                       const float* __restrict__ bias,
                       float* __restrict__ Y, int ldy,
                       _Float16* __restrict__ Yh, int ldyh,
                       int Mout, int nrows, int act) {
  int lane = threadIdx.x & 31;
  int half = lane >> 4, l16 = lane & 15;
  int r0 = blockIdx.x * 16, c0 = blockIdx.y * 16;
  int arow = r0 + l16;
  if (arow >= nrows) arow = nrows - 1;  // clamp: reads stay valid, store guarded
  const _Float16* xr = Xh + (size_t)arow * ldx + (half ? 8 : 0);
  const _Float16* wr = Wt + (size_t)(c0 + l16) * Kp + (half ? 8 : 0);
  v8f acc = {};
  for (int k0 = 0; k0 < Kp; k0 += 32) {
    v8h a0 = *(const v8h*)(xr + k0);
    v8h a1 = *(const v8h*)(xr + k0 + 16);
    v8h b0 = *(const v8h*)(wr + k0);
    v8h b1 = *(const v8h*)(wr + k0 + 16);
    __builtin_prefetch(wr + k0 + 32, 0, 1);
    v16h a, b;
#pragma unroll
    for (int e = 0; e < 8; ++e) {
      a[e] = a0[e]; a[e + 8] = a1[e];
      b[e] = b0[e]; b[e + 8] = b1[e];
    }
    acc = __builtin_amdgcn_wmma_f32_16x16x32_f16(false, a, false, b, (short)0,
                                                 acc, false, false);
  }
  int col = c0 + l16;
  if (col >= Mout) return;
  float bv = bias ? bias[col] : 0.f;
#pragma unroll
  for (int r = 0; r < 8; ++r) {
    int row = r0 + r + (half ? 8 : 0);
    if (row >= nrows) continue;
    float v = acc[r] + bv;
    if (act == 1) v = fmaxf(v, 0.f);
    else if (act == 2) v = v * sigm(v);
    if (Yh) Yh[(size_t)row * ldyh + col] = (_Float16)v;
    else    Y[(size_t)row * ldy + col] = v;
  }
}

// ---- fctp: Y[n,w,iz] += scale * sum_{u,a} X[n,u,iz]*Attr[n,a]*W[u*64+a, w] --
// A fragment = outer product built from LDS-staged tiles; K = mu*64.
__global__ void k_fctp(const float* __restrict__ Xin, int mu, int ncomp,
                       const float* __restrict__ Attr,
                       const _Float16* __restrict__ Wt,
                       float* __restrict__ Y, int mo,
                       int nrows, float scale) {
  __shared__ float sx[16 * 64];
  __shared__ float sa[16 * 64];
  int lane = threadIdx.x & 31;
  int half = lane >> 4, l16 = lane & 15;
  int r0 = blockIdx.x * 16, c0 = blockIdx.y * 16;
  int iz = blockIdx.z;
  for (int t = lane; t < 16 * mu; t += 32) {
    int row = t / mu, u = t - row * mu;
    int g = r0 + row;
    sx[row * 64 + u] =
        (g < nrows) ? Xin[(size_t)g * mu * ncomp + (size_t)u * ncomp + iz] : 0.f;
  }
  for (int t = lane; t < 16 * 64; t += 32) {
    int g = r0 + (t >> 6);
    sa[t] = (g < nrows) ? Attr[(size_t)g * 64 + (t & 63)] : 0.f;
  }
  __syncthreads();
  int Ktot = mu * 64;
  const _Float16* wr = Wt + (size_t)(c0 + l16) * Ktot + (half ? 8 : 0);
  int rbase = l16 * 64;
  v8f acc = {};
  for (int k0 = 0; k0 < Ktot; k0 += 32) {
    v16h a, b;
    int kb = k0 + (half ? 8 : 0);
#pragma unroll
    for (int e = 0; e < 8; ++e) {
      int kk = kb + e;
      a[e] = (_Float16)(sx[rbase + (kk >> 6)] * sa[rbase + (kk & 63)]);
      int kk2 = kb + 16 + e;
      a[e + 8] = (_Float16)(sx[rbase + (kk2 >> 6)] * sa[rbase + (kk2 & 63)]);
    }
    v8h b0 = *(const v8h*)(wr + k0);
    v8h b1 = *(const v8h*)(wr + k0 + 16);
    __builtin_prefetch(wr + k0 + 32, 0, 1);
#pragma unroll
    for (int e = 0; e < 8; ++e) { b[e] = b0[e]; b[e + 8] = b1[e]; }
    acc = __builtin_amdgcn_wmma_f32_16x16x32_f16(false, a, false, b, (short)0,
                                                 acc, false, false);
  }
  int col = c0 + l16;
  if (col >= mo) return;
#pragma unroll
  for (int r = 0; r < 8; ++r) {
    int row = r0 + r + (half ? 8 : 0);
    if (row >= nrows) continue;
    Y[(size_t)row * (mo * ncomp) + (size_t)col * ncomp + iz] += scale * acc[r];
  }
}

// ---- edge geometry: sh = [cut, cut*sqrt(3)*unit] f32; radial basis f16 -----
__global__ void k_edge_geom(const float* __restrict__ evec,
                            float* __restrict__ sh, _Float16* __restrict__ embh,
                            int Ee) {
  int e = blockIdx.x * 256 + threadIdx.x;
  if (e >= Ee) return;
  float x = evec[(size_t)e * 3], y = evec[(size_t)e * 3 + 1],
        z = evec[(size_t)e * 3 + 2];
  float r = sqrtf(x * x + y * y + z * z);
  float rr = fmaxf(r, 1e-12f);
  float ux = x / rr, uy = y / rr, uz = z / rr;
  float u = 2.f * (r / 5.f - 1.f);
  float cut = (u > 0.f) ? 0.f
              : ((u < -1.f) ? 1.f
                            : 0.5f * (1.f - __cosf(3.14159265358979f * u)));
  const float s3 = 1.7320508075688772f;
  sh[(size_t)e * 4 + 0] = cut;
  sh[(size_t)e * 4 + 1] = cut * s3 * ux;
  sh[(size_t)e * 4 + 2] = cut * s3 * uy;
  sh[(size_t)e * 4 + 3] = cut * s3 * uz;
  const float step = 5.f / 9.f;
#pragma unroll
  for (int b = 0; b < 10; ++b) {
    float c = step * b;
    float d = (r - c) / step;
    embh[(size_t)e * 32 + b] =
        (_Float16)(__expf(-d * d) / 1.12f * 3.1622776601683795f);
  }
#pragma unroll
  for (int b = 10; b < 32; ++b) embh[(size_t)e * 32 + b] = (_Float16)0.f;
}

__global__ void k_gather64(const float* __restrict__ src,
                           const int* __restrict__ idx,
                           float* __restrict__ dst, int n) {
  int t = blockIdx.x * 256 + threadIdx.x;
  if (t >= n * 64) return;
  int nd = t >> 6;
  dst[t] = src[(size_t)idx[nd] * 64 + (t & 63)];
}

// ---- conv0 messages: instr {(0,0,0) scalar, (0,1,1) vector}; mu=64, wld=128
__global__ void k_msg0(const int* __restrict__ src, const int* __restrict__ dst,
                       const float* __restrict__ sh, const float* __restrict__ w,
                       const float* __restrict__ x0, float* __restrict__ m0,
                       float* __restrict__ m1, int Ee) {
  int t = blockIdx.x * 256 + threadIdx.x;
  if (t >= Ee * 64) return;
  int e = t >> 6, u = t & 63;
  int s = src[e], d = dst[e];
  const float s8 = 0.35355339059327f;
  float sh0 = sh[(size_t)e * 4], shx = sh[(size_t)e * 4 + 1],
        shy = sh[(size_t)e * 4 + 2], shz = sh[(size_t)e * 4 + 3];
  const float* wr = w + (size_t)e * 128;
  float xv = x0[(size_t)s * 64 + u];
  atomicAdd(&m0[(size_t)d * 64 + u], xv * sh0 * wr[u] * s8);
  float c1 = xv * wr[64 + u] * s8;
  atomicAdd(&m1[(size_t)d * 192 + u * 3 + 0], c1 * shx);
  atomicAdd(&m1[(size_t)d * 192 + u * 3 + 1], c1 * shy);
  atomicAdd(&m1[(size_t)d * 192 + u * 3 + 2], c1 * shz);
}

// ---- conv1 messages: 5 instructions (CG 000,011,101,110,111); mu=32, wld=160
__global__ void k_msg1(const int* __restrict__ src, const int* __restrict__ dst,
                       const float* __restrict__ sh, const float* __restrict__ w,
                       const float* __restrict__ x0, const float* __restrict__ x1,
                       float* __restrict__ m0, float* __restrict__ m1,
                       float* __restrict__ m2, float* __restrict__ m3,
                       float* __restrict__ m4, int Ee) {
  int t = blockIdx.x * 256 + threadIdx.x;
  if (t >= Ee * 32) return;
  int e = t >> 5, u = t & 31;
  int s = src[e], d = dst[e];
  const float s8 = 0.35355339059327f, is3 = 0.57735026918963f,
              is2 = 0.70710678118655f;
  float sh0 = sh[(size_t)e * 4], shx = sh[(size_t)e * 4 + 1],
        shy = sh[(size_t)e * 4 + 2], shz = sh[(size_t)e * 4 + 3];
  const float* wr = w + (size_t)e * 160;
  float xv = x0[(size_t)s * 32 + u];
  const float* xr = x1 + (size_t)s * 96 + u * 3;
  float xa = xr[0], xb = xr[1], xc = xr[2];
  atomicAdd(&m0[(size_t)d * 32 + u], xv * sh0 * wr[u] * s8);
  float c1 = xv * wr[32 + u] * s8;
  atomicAdd(&m1[(size_t)d * 96 + u * 3 + 0], c1 * shx);
  atomicAdd(&m1[(size_t)d * 96 + u * 3 + 1], c1 * shy);
  atomicAdd(&m1[(size_t)d * 96 + u * 3 + 2], c1 * shz);
  float c2 = sh0 * wr[64 + u] * s8;
  atomicAdd(&m2[(size_t)d * 96 + u * 3 + 0], c2 * xa);
  atomicAdd(&m2[(size_t)d * 96 + u * 3 + 1], c2 * xb);
  atomicAdd(&m2[(size_t)d * 96 + u * 3 + 2], c2 * xc);
  atomicAdd(&m3[(size_t)d * 32 + u],
            (xa * shx + xb * shy + xc * shz) * is3 * wr[96 + u] * s8);
  float c4 = is2 * wr[128 + u] * s8;
  atomicAdd(&m4[(size_t)d * 96 + u * 3 + 0], (xb * shz - xc * shy) * c4);
  atomicAdd(&m4[(size_t)d * 96 + u * 3 + 1], (xc * shx - xa * shz) * c4);
  atomicAdd(&m4[(size_t)d * 96 + u * 3 + 2], (xa * shy - xb * shx) * c4);
}

// ---- conv2 messages: instr {(0,0,0), (1,1,->0)}; mu=32, wld=64 -------------
__global__ void k_msg2(const int* __restrict__ src, const int* __restrict__ dst,
                       const float* __restrict__ sh, const float* __restrict__ w,
                       const float* __restrict__ x0, const float* __restrict__ x1,
                       float* __restrict__ m0, float* __restrict__ m1, int Ee) {
  int t = blockIdx.x * 256 + threadIdx.x;
  if (t >= Ee * 32) return;
  int e = t >> 5, u = t & 31;
  int s = src[e], d = dst[e];
  const float s8 = 0.35355339059327f, is3 = 0.57735026918963f;
  float sh0 = sh[(size_t)e * 4], shx = sh[(size_t)e * 4 + 1],
        shy = sh[(size_t)e * 4 + 2], shz = sh[(size_t)e * 4 + 3];
  const float* wr = w + (size_t)e * 64;
  atomicAdd(&m0[(size_t)d * 32 + u], x0[(size_t)s * 32 + u] * sh0 * wr[u] * s8);
  const float* xr = x1 + (size_t)s * 96 + u * 3;
  atomicAdd(&m1[(size_t)d * 32 + u],
            (xr[0] * shx + xr[1] * shy + xr[2] * shz) * is3 * wr[32 + u] * s8);
}

// ---- elementwise helpers ---------------------------------------------------
__global__ void k_silu(const float* __restrict__ A, float* __restrict__ O,
                       int n) {
  int t = blockIdx.x * 256 + threadIdx.x;
  if (t >= n) return;
  float v = A[t];
  O[t] = v * sigm(v);
}
__global__ void k_gate_mul(const float* __restrict__ V,
                           const float* __restrict__ S, float* __restrict__ O,
                           int m, int c, int n) {
  int t = blockIdx.x * 256 + threadIdx.x;
  if (t >= n * m * c) return;
  int nd = t / (m * c);
  int w = (t - nd * m * c) / c;
  O[t] = V[t] * sigm(S[(size_t)nd * m + w]);
}
__global__ void k_axpby(const float* __restrict__ A, const float* __restrict__ B,
                        float* __restrict__ O, float ca, float cb, int n) {
  int t = blockIdx.x * 256 + threadIdx.x;
  if (t >= n) return;
  O[t] = ca * A[t] + cb * B[t];
}
__global__ void k_final(const float* __restrict__ Z, const int* __restrict__ batch,
                        float* __restrict__ out, int n, float scale) {
  int t = blockIdx.x * 256 + threadIdx.x;
  if (t >= n * 4) return;
  int nd = t >> 2;
  atomicAdd(&out[batch[nd] * 4 + (t & 3)], Z[t] * scale);
}

// ---------------------------------------------------------------------------
// Host orchestration. d_in flattening assumed to follow setup_inputs() dict
// insertion order, nested params depth-first:
//  0..7  tensors; 8..15 em_{w,b}, em_attr_{w,b}, em_crystal_{w,b}, em_crys_{w,b}
//  conv0: 16,17 sc_attr(0_0,0_1) 18,19 sc_crys 20 l1_attr 21 l1_crys
//         22,23 l2_attr(0_0,1_1) 24,25 l2_crys 26,27,28 l3(0_0,0_1,1_2)
//         29 fc0 30 fc1
//  conv1: 31..34 sc_attr(0_0,0_1,0_2,1_3) 35..38 sc_crys 39,40 l1_attr
//         41,42 l1_crys 43..51 l2_attr(0_0,3_0,1_1,2_1,1_2,2_2,0_3,3_3,4_4)
//         52..60 l2_crys 61..69 l3(0_0,3_0,0_1,3_1,0_2,3_2,1_3,2_3,4_4)
//         70 fc0 71 fc1
//  conv2: 72 sc_attr 73 sc_crys 74..76 l1_attr 77..79 l1_crys
//         80..83 l2_attr(0_0,1_0,0_1,1_1) 84..87 l2_crys 88,89 l3(0_0,1_0)
//         90 fc0 91 fc1
// ---------------------------------------------------------------------------
extern "C" void kernel_launch(void* const* d_in, const int* in_sizes, int n_in,
                              void* d_out, int out_size, void* d_ws,
                              size_t ws_size, hipStream_t stream) {
  (void)n_in; (void)out_size; (void)ws_size;
  const int Nn = in_sizes[0] / 92;
  const int Ee = in_sizes[3];
  const int* esrc = (const int*)d_in[3];
  const int* edst = (const int*)d_in[4];
  const int* batch = (const int*)d_in[7];

  char* base = (char*)d_ws;
  size_t off = 0;
  auto alloc = [&](size_t bytes) -> void* {
    off = (off + 255) & ~(size_t)255;
    void* p = base + off;
    off += bytes;
    return p;
  };

  // ---- cast all weight matrices to f16 transposed [Mp][Kp] ----
  struct WT { _Float16* p; int Kp, Mp; };
  WT wt[92] = {};
  auto castw = [&](int idx, int K, int M) {
    int Kp = (K + 31) & ~31, Mp = (M + 15) & ~15;
    wt[idx].p = (_Float16*)alloc((size_t)Kp * Mp * 2);
    wt[idx].Kp = Kp;
    wt[idx].Mp = Mp;
    int tot = Kp * Mp;
    k_cast_wt<<<(tot + 255) / 256, 256, 0, stream>>>(
        (const float*)d_in[idx], wt[idx].p, K, M, Kp, Mp);
  };
  castw(8, 92, 64); castw(10, 92, 64); castw(12, 144, 64); castw(14, 48, 64);
  for (int i = 16; i <= 19; ++i) castw(i, 4096, 32);
  for (int i = 20; i <= 25; ++i) castw(i, 4096, 64);
  for (int i = 26; i <= 28; ++i) castw(i, 4096, 32);
  castw(29, 10, 100); castw(30, 100, 128);
  for (int i = 31; i <= 69; ++i) castw(i, 2048, 32);
  castw(70, 10, 100); castw(71, 100, 160);
  castw(72, 2048, 4); castw(73, 2048, 4);
  for (int i = 74; i <= 87; ++i) castw(i, 2048, 32);
  castw(88, 2048, 4); castw(89, 2048, 4);
  castw(90, 10, 100); castw(91, 100, 64);

  // ---- buffers ----
  _Float16* nodeh = (_Float16*)alloc((size_t)Nn * 96 * 2);
  _Float16* attrh = (_Float16*)alloc((size_t)Nn * 96 * 2);
  _Float16* crysh = (_Float16*)alloc((size_t)Nn * 160 * 2);
  _Float16* symh  = (_Float16*)alloc((size_t)16 * 64 * 2);
  _Float16* embh  = (_Float16*)alloc((size_t)Ee * 32 * 2);
  _Float16* hidh  = (_Float16*)alloc((size_t)Ee * 128 * 2);
  float* h    = (float*)alloc((size_t)Nn * 64 * 4);
  float* na   = (float*)alloc((size_t)Nn * 64 * 4);
  float* ncb  = (float*)alloc((size_t)Nn * 64 * 4);
  float* sym  = (float*)alloc((size_t)16 * 64 * 4);
  float* symN = (float*)alloc((size_t)Nn * 64 * 4);
  float* sh   = (float*)alloc((size_t)Ee * 4 * 4);
  float* wbuf = (float*)alloc((size_t)Ee * 160 * 4);
  float* slab = (float*)alloc((size_t)Nn * 1408 * 4);  // per-conv scratch
  float* gA   = (float*)alloc((size_t)Nn * 128 * 4);   // conv0 gated out
  float* gB   = (float*)alloc((size_t)Nn * 224 * 4);   // conv1 gated out

  auto gemm = [&](const _Float16* Xh, int ldx, int widx, const float* bias,
                  float* Y, int ldy, _Float16* Yh, int ldyh, int Mout,
                  int nrows, int act) {
    dim3 g((nrows + 15) / 16, wt[widx].Mp / 16);
    k_gemm<<<g, 32, 0, stream>>>(Xh, ldx, wt[widx].p, wt[widx].Kp, bias, Y, ldy,
                                 Yh, ldyh, Mout, nrows, act);
  };
  auto fctp = [&](const float* X, int mu, int ncomp, const float* attr,
                  int widx, float* Y, int mo, int npaths) {
    float scale = 1.f / sqrtf((float)mu * 64.f * (float)npaths);
    dim3 g((Nn + 15) / 16, (mo + 15) / 16, ncomp);
    k_fctp<<<g, 32, 0, stream>>>(X, mu, ncomp, attr, wt[widx].p, Y, mo, Nn,
                                 scale);
  };
  auto axpby = [&](const float* A, const float* B, float* O, float ca,
                   float cb, int n) {
    k_axpby<<<(n + 255) / 256, 256, 0, stream>>>(A, B, O, ca, cb, n);
  };
  const float cs = 0.3826834323650898f, cx = 0.9238795325112867f;

  // ---- input casts (zero-padded f16 GEMM operands) ----
  k_cast_xh<<<(Nn * 96 + 255) / 256, 256, 0, stream>>>(
      (const float*)d_in[0], 92, 92, nodeh, 96, Nn);
  k_cast_xh<<<(Nn * 96 + 255) / 256, 256, 0, stream>>>(
      (const float*)d_in[1], 92, 92, attrh, 96, Nn);
  k_cast_xh<<<(Nn * 160 + 255) / 256, 256, 0, stream>>>(
      (const float*)d_in[2], 144, 144, crysh, 160, Nn);
  k_cast_xh<<<(16 * 64 + 255) / 256, 256, 0, stream>>>(
      (const float*)d_in[6], 48, 48, symh, 64, 16);
  hipMemsetAsync(hidh, 0, (size_t)Ee * 128 * 2, stream);  // keep pad cols zero

  // ---- embeddings ----
  gemm(nodeh, 96, 8, (const float*)d_in[9], h, 64, nullptr, 0, 64, Nn, 1);
  gemm(attrh, 96, 10, (const float*)d_in[11], na, 64, nullptr, 0, 64, Nn, 1);
  gemm(crysh, 160, 12, (const float*)d_in[13], ncb, 64, nullptr, 0, 64, Nn, 1);
  gemm(symh, 64, 14, (const float*)d_in[15], sym, 64, nullptr, 0, 64, 16, 1);
  k_gather64<<<(Nn * 64 + 255) / 256, 256, 0, stream>>>(sym, batch, symN, Nn);
  k_edge_geom<<<(Ee + 255) / 256, 256, 0, stream>>>((const float*)d_in[5], sh,
                                                    embh, Ee);

  // ================= conv0: IRR_EM -> C1_OUT =================
  {
    gemm(embh, 32, 29, nullptr, nullptr, 0, hidh, 128, 112, Ee, 2);
    gemm(hidh, 128, 30, nullptr, wbuf, 128, nullptr, 0, 128, Ee, 0);
    float* s0 = slab;
    float* s1 = s0 + (size_t)Nn * 32;
    float* s2 = s1 + (size_t)Nn * 32;
    float* x1 = s2 + (size_t)Nn * 96;
    float* m0 = x1 + (size_t)Nn * 64;
    float* m1 = m0 + (size_t)Nn * 64;
    float* y0 = m1 + (size_t)Nn * 192;
    float* y1 = y0 + (size_t)Nn * 64;
    float* z0 = y1 + (size_t)Nn * 192;
    float* z1 = z0 + (size_t)Nn * 32;
    float* z2 = z1 + (size_t)Nn * 32;
    hipMemsetAsync(slab, 0, (size_t)Nn * 896 * 4, stream);
    fctp(h, 64, 1, na, 16, s0, 32, 1);  fctp(h, 64, 1, ncb, 18, s0, 32, 1);
    fctp(h, 64, 1, na, 17, s1, 32, 1);  fctp(h, 64, 1, ncb, 19, s1, 32, 1);
    fctp(h, 64, 1, na, 20, x1, 64, 1);  fctp(h, 64, 1, ncb, 21, x1, 64, 1);
    k_msg0<<<(Ee * 64 + 255) / 256, 256, 0, stream>>>(esrc, edst, sh, wbuf, x1,
                                                      m0, m1, Ee);
    fctp(m0, 64, 1, na, 22, y0, 64, 1); fctp(m0, 64, 1, ncb, 24, y0, 64, 1);
    fctp(m1, 64, 3, na, 23, y1, 64, 1); fctp(m1, 64, 3, ncb, 25, y1, 64, 1);
    fctp(y0, 64, 1, symN, 26, z0, 32, 1);
    fctp(y0, 64, 1, symN, 27, z1, 32, 1);
    fctp(y1, 64, 3, symN, 28, z2, 32, 1);
    axpby(s0, z0, z0, cs, cx, Nn * 32);
    axpby(s1, z1, z1, cs, cx, Nn * 32);
    axpby(s2, z2, z2, cs, 1.f, Nn * 96);
    k_silu<<<(Nn * 32 + 255) / 256, 256, 0, stream>>>(z0, gA, Nn * 32);
    k_gate_mul<<<(Nn * 96 + 255) / 256, 256, 0, stream>>>(
        z2, z1, gA + (size_t)Nn * 32, 32, 3, Nn);
  }

  // ================= conv1: L1_OUT -> C2_OUT =================
  {
    gemm(embh, 32, 70, nullptr, nullptr, 0, hidh, 128, 112, Ee, 2);
    gemm(hidh, 128, 71, nullptr, wbuf, 160, nullptr, 0, 160, Ee, 0);
    float* g0 = gA;
    float* g1 = gA + (size_t)Nn * 32;
    float* s0 = slab;
    float* s1 = s0 + (size_t)Nn * 32;
    float* s2 = s1 + (size_t)Nn * 32;
    float* s3 = s2 + (size_t)Nn * 32;
    float* s4 = s3 + (size_t)Nn * 96;  // (1,1) block stays zero (no path)
    float* x0 = s4 + (size_t)Nn * 96;
    float* x1 = x0 + (size_t)Nn * 32;
    float* m0 = x1 + (size_t)Nn * 96;
    float* m1 = m0 + (size_t)Nn * 32;
    float* m2 = m1 + (size_t)Nn * 96;
    float* m3 = m2 + (size_t)Nn * 96;
    float* m4 = m3 + (size_t)Nn * 32;
    float* y0 = m4 + (size_t)Nn * 96;
    float* y1 = y0 + (size_t)Nn * 32;
    float* y2 = y1 + (size_t)Nn * 96;
    float* y3 = y2 + (size_t)Nn * 96;
    float* y4 = y3 + (size_t)Nn * 32;
    float* z0 = y4 + (size_t)Nn * 96;
    float* z1 = z0 + (size_t)Nn * 32;
    float* z2 = z1 + (size_t)Nn * 32;
    float* z3 = z2 + (size_t)Nn * 32;
    float* z4 = z3 + (size_t)Nn * 96;
    hipMemsetAsync(slab, 0, (size_t)Nn * 1408 * 4, stream);
    // self-connection
    fctp(g0, 32, 1, na, 31, s0, 32, 1); fctp(g0, 32, 1, ncb, 35, s0, 32, 1);
    fctp(g0, 32, 1, na, 32, s1, 32, 1); fctp(g0, 32, 1, ncb, 36, s1, 32, 1);
    fctp(g0, 32, 1, na, 33, s2, 32, 1); fctp(g0, 32, 1, ncb, 37, s2, 32, 1);
    fctp(g1, 32, 3, na, 34, s3, 32, 1); fctp(g1, 32, 3, ncb, 38, s3, 32, 1);
    // l1
    fctp(g0, 32, 1, na, 39, x0, 32, 1); fctp(g0, 32, 1, ncb, 41, x0, 32, 1);
    fctp(g1, 32, 3, na, 40, x1, 32, 1); fctp(g1, 32, 3, ncb, 42, x1, 32, 1);
    // messages
    k_msg1<<<(Ee * 32 + 255) / 256, 256, 0, stream>>>(esrc, edst, sh, wbuf, x0,
                                                      x1, m0, m1, m2, m3, m4, Ee);
    // l2 (mid -> mid)
    fctp(m0, 32, 1, na, 43, y0, 32, 2); fctp(m3, 32, 1, na, 44, y0, 32, 2);
    fctp(m0, 32, 1, ncb, 52, y0, 32, 2); fctp(m3, 32, 1, ncb, 53, y0, 32, 2);
    fctp(m1, 32, 3, na, 45, y1, 32, 2); fctp(m2, 32, 3, na, 46, y1, 32, 2);
    fctp(m1, 32, 3, ncb, 54, y1, 32, 2); fctp(m2, 32, 3, ncb, 55, y1, 32, 2);
    fctp(m1, 32, 3, na, 47, y2, 32, 2); fctp(m2, 32, 3, na, 48, y2, 32, 2);
    fctp(m1, 32, 3, ncb, 56, y2, 32, 2); fctp(m2, 32, 3, ncb, 57, y2, 32, 2);
    fctp(m0, 32, 1, na, 49, y3, 32, 2); fctp(m3, 32, 1, na, 50, y3, 32, 2);
    fctp(m0, 32, 1, ncb, 58, y3, 32, 2); fctp(m3, 32, 1, ncb, 59, y3, 32, 2);
    fctp(m4, 32, 3, na, 51, y4, 32, 1); fctp(m4, 32, 3, ncb, 60, y4, 32, 1);
    // l3 (mid -> C2_OUT, attr = sym_nodes)
    fctp(y0, 32, 1, symN, 61, z0, 32, 2); fctp(y3, 32, 1, symN, 62, z0, 32, 2);
    fctp(y0, 32, 1, symN, 63, z1, 32, 2); fctp(y3, 32, 1, symN, 64, z1, 32, 2);
    fctp(y0, 32, 1, symN, 65, z2, 32, 2); fctp(y3, 32, 1, symN, 66, z2, 32, 2);
    fctp(y1, 32, 3, symN, 67, z3, 32, 2); fctp(y2, 32, 3, symN, 68, z3, 32, 2);
    fctp(y4, 32, 3, symN, 69, z4, 32, 1);
    // combine + gate(1,2)
    axpby(s0, z0, z0, cs, cx, Nn * 32);
    axpby(s1, z1, z1, cs, cx, Nn * 32);
    axpby(s2, z2, z2, cs, cx, Nn * 32);
    axpby(s3, z3, z3, cs, cx, Nn * 96);
    axpby(s4, z4, z4, cs, 1.f, Nn * 96);
    k_silu<<<(Nn * 32 + 255) / 256, 256, 0, stream>>>(z0, gB, Nn * 32);
    k_gate_mul<<<(Nn * 96 + 255) / 256, 256, 0, stream>>>(
        z3, z1, gB + (size_t)Nn * 32, 32, 3, Nn);
    k_gate_mul<<<(Nn * 96 + 255) / 256, 256, 0, stream>>>(
        z4, z2, gB + (size_t)Nn * 128, 32, 3, Nn);
  }

  // ================= conv2: L2_OUT -> IRR_OUT ================
  {
    gemm(embh, 32, 90, nullptr, nullptr, 0, hidh, 128, 112, Ee, 2);
    gemm(hidh, 128, 91, nullptr, wbuf, 64, nullptr, 0, 64, Ee, 0);
    float* g0 = gB;
    float* g1 = gB + (size_t)Nn * 32;  // (1,1) input block unused by instrs
    float* s  = slab;
    float* x0 = s + (size_t)Nn * 16;
    float* x1 = x0 + (size_t)Nn * 32;
    float* m0 = x1 + (size_t)Nn * 96;
    float* m1 = m0 + (size_t)Nn * 32;
    float* y0 = m1 + (size_t)Nn * 32;
    float* y1 = y0 + (size_t)Nn * 32;
    float* z  = y1 + (size_t)Nn * 32;
    hipMemsetAsync(slab, 0, (size_t)Nn * 288 * 4, stream);
    fctp(g0, 32, 1, na, 72, s, 4, 1);  fctp(g0, 32, 1, ncb, 73, s, 4, 1);
    fctp(g0, 32, 1, na, 74, x0, 32, 1); fctp(g0, 32, 1, ncb, 77, x0, 32, 1);
    fctp(g1, 32, 3, na, 75, x1, 32, 1); fctp(g1, 32, 3, ncb, 78, x1, 32, 1);
    k_msg2<<<(Ee * 32 + 255) / 256, 256, 0, stream>>>(esrc, edst, sh, wbuf, x0,
                                                      x1, m0, m1, Ee);
    fctp(m0, 32, 1, na, 80, y0, 32, 2);  fctp(m1, 32, 1, na, 81, y0, 32, 2);
    fctp(m0, 32, 1, ncb, 84, y0, 32, 2); fctp(m1, 32, 1, ncb, 85, y0, 32, 2);
    fctp(m0, 32, 1, na, 82, y1, 32, 2);  fctp(m1, 32, 1, na, 83, y1, 32, 2);
    fctp(m0, 32, 1, ncb, 86, y1, 32, 2); fctp(m1, 32, 1, ncb, 87, y1, 32, 2);
    fctp(y0, 32, 1, symN, 88, z, 4, 2);
    fctp(y1, 32, 1, symN, 89, z, 4, 2);
    axpby(s, z, z, cs, cx, Nn * 4);
    hipMemsetAsync(d_out, 0, 64 * 4, stream);
    k_final<<<(Nn * 4 + 255) / 256, 256, 0, stream>>>(
        z, batch, (float*)d_out, Nn, 1.f / sqrtf((float)Nn / 16.f));
  }
}
